// MultiHeadAttention_44959717654813
// MI455X (gfx1250) — matrix operand order repaired and checked by reading, not compiled, not run on hardware
//
#include <hip/hip_runtime.h>
#include <math.h>

// ---------------------------------------------------------------------------
// MHA forward for MI455X (gfx1250, wave32, WMMA).
// Shapes: B=4, S=1024, H=16, D=64, DM=1024.
// d_out = [ out (4M f32) | attn (64M f32) ].
// Workspace (bf16): q/k/v casts (3x4M), W casts (4x1M), Qp/Kp/Vp (3x4M), ctx (4M).
// ---------------------------------------------------------------------------

typedef __bf16 bf16;
typedef __attribute__((ext_vector_type(16))) __bf16 v16bf;
typedef __attribute__((ext_vector_type(8)))  __bf16 v8bf;
typedef __attribute__((ext_vector_type(8)))  float  v8f;
typedef __attribute__((ext_vector_type(4)))  int    v4i;

#define B_  4
#define S_  1024
#define H_  16
#define D_  64
#define DM_ 1024
#define NROW_ (B_ * S_)
#define INV_SCALE 0.125f
#define OUT_ELEMS ((size_t)B_ * S_ * DM_)
#define VT_LD 1040   // padded V^T row stride (elements) to spread LDS banks

// ---- CDNA5 async global->LDS copy (guarded; falls back to sync copy) ------
#if __has_builtin(__builtin_amdgcn_global_load_async_to_lds_b128)
#define ASYNC_COPY_B128(gsrc, ldst)                                          \
  __builtin_amdgcn_global_load_async_to_lds_b128((v4i*)(gsrc),               \
                                                 (v4i*)(ldst), 0, 0)
#if __has_builtin(__builtin_amdgcn_s_wait_asynccnt)
#define ASYNC_WAIT() __builtin_amdgcn_s_wait_asynccnt(0)
#else
#define ASYNC_WAIT() asm volatile("s_wait_asynccnt 0x0" ::: "memory")
#endif
#else
#define ASYNC_COPY_B128(gsrc, ldst) (*(v8bf*)(ldst) = *(const v8bf*)(gsrc))
#define ASYNC_WAIT()
#endif

union V16u { v16bf v; v8bf h[2]; bf16 e[16]; };

// ---- WMMA fragment loaders (contiguous k order: lane half h holds k = 16h..16h+15) ------

// A 16x32 bf16 from row-major [rows x ld].
template <typename P>
__device__ inline v16bf load_a_frag(P base, int ld, int row0, int kk, int lane) {
  int r  = row0 + (lane & 15);
  int kb = (lane < 16) ? 0 : 16;
  P p = base + (size_t)r * ld + kk + kb;
  V16u u;
  u.h[0] = *(const v8bf*)(p);
  u.h[1] = *(const v8bf*)(p + 8);
  return u.v;
}

// B 32x16 bf16 where B[k][n] = src[n*ld + k]  (k contiguous per lane).
template <typename P>
__device__ inline v16bf load_b_frag_rowk(P base, int ld, int n0, int kk, int lane) {
  int n  = n0 + (lane & 15);
  int kb = kk + ((lane < 16) ? 0 : 16);
  return *(const v16bf*)(base + (size_t)n * ld + kb);
}

__device__ inline v8f wmma_bf16(v16bf a, v16bf b, v8f c) {
  return __builtin_amdgcn_wmma_f32_16x16x32_bf16(
      false, a, false, b, (short)0, c, false, false);
}

// ---------------------------------------------------------------------------
// Kernel 0: f32 -> bf16 cast
// ---------------------------------------------------------------------------
__global__ void k_cast_bf16(const float* __restrict__ in, bf16* __restrict__ out,
                            int n) {
  int i = blockIdx.x * blockDim.x + threadIdx.x;
  if (i < n) out[i] = (bf16)in[i];
}

// ---------------------------------------------------------------------------
// Shared GEMM core: Y = X @ W^T with LDS-staged, double-buffered W strip.
// Block = 8 waves -> 128 rows x 64 cols. Grid = (DM/64, NROW/128).
// ---------------------------------------------------------------------------
template <bool HEADS_OUT, typename OUT_T>
__device__ inline void gemm_xwT(const bf16* __restrict__ X,
                                const bf16* __restrict__ W,
                                OUT_T* __restrict__ Y) {
  __shared__ bf16 Wt[2][64 * 32];          // double-buffered 64x32 strip (8KB)

  const int tid  = threadIdx.x;
  const int lane = tid & 31;
  const int wv   = tid >> 5;
  const int row0 = (blockIdx.y * 8 + wv) * 16;
  const int n0   = blockIdx.x * 64;

  // per-thread staging slot: col sn (0..63), k-offset sk (0,8,16,24)
  const int sn = tid >> 2;
  const int sk = (tid & 3) * 8;
  const bf16* wsrc = W + (size_t)(n0 + sn) * DM_ + sk;
  bf16* wdst[2] = { &Wt[0][sn * 32 + sk], &Wt[1][sn * 32 + sk] };

  ASYNC_COPY_B128(wsrc, wdst[0]);
  ASYNC_WAIT();
  __syncthreads();

  const int cl   = lane & 15;
  const int kb16 = (lane < 16) ? 0 : 16;

  v8f acc[4] = {};
  int buf = 0;
  for (int kk = 0; kk < DM_; kk += 32) {
    if (kk + 32 < DM_)
      ASYNC_COPY_B128(wsrc + kk + 32, wdst[buf ^ 1]);    // prefetch next chunk

    v16bf a = load_a_frag(X, DM_, row0, kk, lane);
    __builtin_prefetch(X + (size_t)(row0 + cl) * DM_ + kk + 64, 0, 1);
    const bf16* wl = &Wt[buf][0];
#pragma unroll
    for (int t = 0; t < 4; ++t) {
      v16bf b = *(const v16bf*)&wl[(t * 16 + cl) * 32 + kb16];
      acc[t] = wmma_bf16(a, b, acc[t]);
    }
    ASYNC_WAIT();
    __syncthreads();
    buf ^= 1;
  }

  const int rbase = row0 + ((lane < 16) ? 0 : 8);
#pragma unroll
  for (int t = 0; t < 4; ++t) {
#pragma unroll
    for (int r = 0; r < 8; ++r) {
      int m = rbase + r;
      int n = n0 + t * 16 + cl;
      if (HEADS_OUT) {   // scatter bf16 into (B,H,S,D)
        int b = m >> 10, s = m & 1023, h = n >> 6, d = n & 63;
        ((bf16*)Y)[(((size_t)b * H_ + h) * S_ + s) * D_ + d] = (bf16)acc[t][r];
      } else {           // plain f32 row-major, non-temporal
        __builtin_nontemporal_store(acc[t][r], (float*)Y + (size_t)m * DM_ + n);
      }
    }
  }
}

__global__ void __launch_bounds__(256)
k_proj_gemm_heads(const bf16* __restrict__ X, const bf16* __restrict__ W,
                  bf16* __restrict__ Y) {
  gemm_xwT<true>(X, W, Y);
}

__global__ void __launch_bounds__(256)
k_out_gemm(const bf16* __restrict__ X, const bf16* __restrict__ W,
           float* __restrict__ Y) {
  gemm_xwT<false>(X, W, Y);
}

// ---------------------------------------------------------------------------
// Kernel 2: fused attention for one (b,h, 16-row q tile).
// grid = (S/16, B*H), block = 256 (8 waves), dynamic LDS (~231KB of 320KB WGP).
// ---------------------------------------------------------------------------
__global__ void __launch_bounds__(256)
k_attention(const bf16* __restrict__ Qp, const bf16* __restrict__ Kp,
            const bf16* __restrict__ Vp, float* __restrict__ attn_out,
            bf16* __restrict__ ctx_out) {
  extern __shared__ char smem[];
  float* scores = (float*)smem;                                   // 16x1024 f32
  bf16*  probs  = (bf16*)(smem + 16 * 1024 * 4);                  // 16x1024 bf16
  bf16*  vt     = (bf16*)(smem + 16 * 1024 * 4 + 16 * 1024 * 2);  // 64 x VT_LD
  float* ctxs   = (float*)((char*)vt + D_ * VT_LD * 2);           // 16x64 f32
  float* red    = ctxs + 16 * 64;                                 // 256
  float* red2   = red + 256;                                      // 16

  const int tid  = threadIdx.x;
  const int lane = tid & 31;
  const int wv   = tid >> 5;
  const int q0   = blockIdx.x * 16;
  const int bh   = blockIdx.y;          // b*H + h

  const bf16* Qh = Qp + (size_t)bh * S_ * D_;
  const bf16* Kh = Kp + (size_t)bh * S_ * D_;
  const bf16* Vh = Vp + (size_t)bh * S_ * D_;

  for (int i = tid; i < 16 * 64; i += 256) ctxs[i] = 0.0f;

  // ---- stage V^T into LDS: vt[d][k], coalesced global reads ---------------
#pragma unroll 4
  for (int i = 0; i < 32; ++i) {
    int f = (i * 256 + tid) * 8;        // 8 consecutive d at row k
    int k = f >> 6, d = f & 63;
    v8bf x = *(const v8bf*)(Vh + f);
#pragma unroll
    for (int j = 0; j < 8; ++j) vt[(d + j) * VT_LD + k] = x[j];
  }

  // ---- scores = (Q K^T)/sqrt(D): wave w covers k-tiles [8w, 8w+8) ---------
  v16bf a0 = load_a_frag(Qh, D_, q0, 0, lane);
  v16bf a1 = load_a_frag(Qh, D_, q0, 32, lane);
  const int rb = (lane < 16) ? 0 : 8;
  const int cl = lane & 15;
#pragma unroll
  for (int j = 0; j < 8; ++j) {
    int kt = wv * 8 + j;
    v8f c = {};
    v16bf b0 = load_b_frag_rowk(Kh, D_, kt * 16, 0, lane);   // B[d,k]=K[k,d]
    c = wmma_bf16(a0, b0, c);
    v16bf b1 = load_b_frag_rowk(Kh, D_, kt * 16, 32, lane);
    c = wmma_bf16(a1, b1, c);
#pragma unroll
    for (int r = 0; r < 8; ++r)
      scores[(rb + r) * 1024 + kt * 16 + cl] = c[r] * INV_SCALE;
  }
  __syncthreads();

  // ---- softmax: 16 threads x 64 elems per row -----------------------------
  const int row = tid >> 4, seg = tid & 15;
  float* srow = scores + row * 1024 + seg * 64;
  float mx = -INFINITY;
#pragma unroll 4
  for (int i = 0; i < 64; ++i) mx = fmaxf(mx, srow[i]);
  red[row * 16 + seg] = mx;
  __syncthreads();
  if (seg == 0) {
    float m = red[row * 16];
    for (int i = 1; i < 16; ++i) m = fmaxf(m, red[row * 16 + i]);
    red2[row] = m;
  }
  __syncthreads();
  const float rm = red2[row];
  float sum = 0.0f;
#pragma unroll 4
  for (int i = 0; i < 64; ++i) {
    float e = __expf(srow[i] - rm);
    srow[i] = e;
    sum += e;
  }
  red[row * 16 + seg] = sum;
  __syncthreads();
  if (seg == 0) {
    float s = 0.0f;
    for (int i = 0; i < 16; ++i) s += red[row * 16 + i];
    red2[row] = 1.0f / s;
  }
  __syncthreads();
  const float inv = red2[row];
  float* arow = attn_out + ((size_t)bh * S_ + q0 + row) * S_ + seg * 64;
#pragma unroll 4
  for (int i = 0; i < 64; ++i) {
    float p = srow[i] * inv;
    __builtin_nontemporal_store(p, arow + i);      // 256MB stream -> NT
    probs[row * 1024 + seg * 64 + i] = (bf16)p;
  }
  __syncthreads();

  // ---- ctx = P @ V : 4 d-tiles x 2 split-K halves across 8 waves ----------
  const int dt = wv & 3;
  const int kh = wv >> 2;
  v8f c = {};
  for (int kc = 0; kc < 16; ++kc) {
    int kk = kh * 512 + kc * 32;
    v16bf a = load_a_frag((const bf16*)probs, 1024, 0, kk, lane);
    v16bf b = load_b_frag_rowk((const bf16*)vt, VT_LD, dt * 16, kk, lane);
    c = wmma_bf16(a, b, c);
  }
#pragma unroll
  for (int r = 0; r < 8; ++r)
    atomicAdd(&ctxs[(rb + r) * 64 + dt * 16 + cl], c[r]);   // ds_add_f32
  __syncthreads();

  // ---- write ctx tile (b,s,h,d) row-major ---------------------------------
  const int bb = bh >> 4, hh = bh & 15;
  for (int i = tid; i < 16 * 64; i += 256) {
    int qr = i >> 6, d = i & 63;
    ctx_out[(((size_t)bb * S_ + q0 + qr) * H_ + hh) * D_ + d] = (bf16)ctxs[i];
  }
}

// ---------------------------------------------------------------------------
extern "C" void kernel_launch(void* const* d_in, const int* in_sizes, int n_in,
                              void* d_out, int out_size, void* d_ws, size_t ws_size,
                              hipStream_t stream) {
  const float* query = (const float*)d_in[0];
  const float* key   = (const float*)d_in[1];
  const float* value = (const float*)d_in[2];
  const float* Wq    = (const float*)d_in[3];
  const float* Wk    = (const float*)d_in[4];
  const float* Wv    = (const float*)d_in[5];
  const float* Wo    = (const float*)d_in[6];

  float* out  = (float*)d_out;
  float* attn = out + OUT_ELEMS;

  bf16* ws  = (bf16*)d_ws;
  bf16* qx  = ws;
  bf16* kx  = qx + OUT_ELEMS;
  bf16* vx  = kx + OUT_ELEMS;
  bf16* wqb = vx + OUT_ELEMS;
  bf16* wkb = wqb + DM_ * DM_;
  bf16* wvb = wkb + DM_ * DM_;
  bf16* wob = wvb + DM_ * DM_;
  bf16* Qp  = wob + DM_ * DM_;
  bf16* Kp  = Qp + OUT_ELEMS;
  bf16* Vp  = Kp + OUT_ELEMS;
  bf16* ctx = Vp + OUT_ELEMS;

  const int nAct = B_ * S_ * DM_;
  const int nWgt = DM_ * DM_;

  k_cast_bf16<<<nAct / 256, 256, 0, stream>>>(query, qx, nAct);
  k_cast_bf16<<<nAct / 256, 256, 0, stream>>>(key,   kx, nAct);
  k_cast_bf16<<<nAct / 256, 256, 0, stream>>>(value, vx, nAct);
  k_cast_bf16<<<nWgt / 256, 256, 0, stream>>>(Wq, wqb, nWgt);
  k_cast_bf16<<<nWgt / 256, 256, 0, stream>>>(Wk, wkb, nWgt);
  k_cast_bf16<<<nWgt / 256, 256, 0, stream>>>(Wv, wvb, nWgt);
  k_cast_bf16<<<nWgt / 256, 256, 0, stream>>>(Wo, wob, nWgt);

  dim3 gGemm(DM_ / 64, NROW_ / 128);   // (16, 32)
  k_proj_gemm_heads<<<gGemm, 256, 0, stream>>>(qx, wqb, Qp);
  k_proj_gemm_heads<<<gGemm, 256, 0, stream>>>(kx, wkb, Kp);
  k_proj_gemm_heads<<<gGemm, 256, 0, stream>>>(vx, wvb, Vp);

  size_t lds = 16 * 1024 * 4          // scores
             + 16 * 1024 * 2          // probs
             + (size_t)D_ * VT_LD * 2 // V^T
             + 16 * 64 * 4            // ctx
             + (256 + 16) * 4;        // reductions
  dim3 gAttn(S_ / 16, B_ * H_);
  k_attention<<<gAttn, 256, lds, stream>>>(Qp, Kp, Vp, attn, ctx);

  k_out_gemm<<<gGemm, 256, 0, stream>>>(ctx, wob, out);
}
